// DeformationNetworkSeparate_44976897524293
// MI455X (gfx1250) — matrix operand order repaired and checked by reading, not compiled, not run on hardware
//
#include <hip/hip_runtime.h>
#include <hip/hip_bf16.h>

typedef _Float16 f16;
typedef __attribute__((ext_vector_type(16))) _Float16 v16h;
typedef __attribute__((ext_vector_type(8)))  _Float16 v8h;
typedef __attribute__((ext_vector_type(8)))  float    v8f;

#define MAXL 10

struct LayerDesc { int dinP; int doutP; unsigned wOff; unsigned bOff; };
struct NetDesc   { int nLayers; LayerDesc L[MAXL]; };

// ---------------------------------------------------------------------------
// Prep: convert one layer's f32 W[dout][din] -> fp16 Wp[doutP][dinP] (zero
// padded) and b[dout] -> f32 bp[doutP]. Zero padding guarantees padded
// activation columns stay exactly 0 through ReLU layers.
// ---------------------------------------------------------------------------
__global__ void pack_layer_kernel(const float* __restrict__ W,
                                  const float* __restrict__ b,
                                  f16* __restrict__ Wp,
                                  float* __restrict__ bp,
                                  int dout, int din, int doutP, int dinP)
{
    const int total = doutP * dinP;
    const int idx = blockIdx.x * 256 + threadIdx.x;
    if (idx < total) {
        const int o = idx / dinP;
        const int i = idx - o * dinP;
        float v = (o < dout && i < din) ? W[(size_t)o * din + i] : 0.0f;
        Wp[idx] = (f16)v;
    } else {
        const int j = idx - total;
        if (j < doutP) bp[j] = (j < dout) ? b[j] : 0.0f;
    }
}

// ---------------------------------------------------------------------------
// Process NT output column tiles (spaced 2 tiles apart) with NT independent
// WMMA accumulator chains. One A-fragment LDS load is shared by NT WMMAs per
// K-step, and the NT chains interleave on the matrix pipe.
//   A frag:           lane(hi,rlow), elem j -> K = k0 + 16*hi + j  (16 contiguous halves)
//   B frag:           elem j = W[col][k0 + 16*hi + j]  (16 contiguous halves)
//   D frag:           elem e -> row e + 8*hi, col = rlow (+ tile*16)
// ---------------------------------------------------------------------------
template<int NT>
__device__ __forceinline__ void mma_group(const f16* __restrict__ aRow,
                                          const f16* __restrict__ W,
                                          const float* __restrict__ Bias,
                                          f16* __restrict__ nxt,
                                          int dinP, int doutP,
                                          int ctBase, int rowBase,
                                          int hi, int rlow, bool last)
{
    v8f acc[NT];
    const f16* wRow[NT];
    #pragma unroll
    for (int tn = 0; tn < NT; ++tn) {
        acc[tn] = (v8f){};
        wRow[tn] = W + (size_t)((ctBase + 2 * tn) * 16 + rlow) * dinP + hi * 16;
    }

    for (int k0 = 0; k0 < dinP; k0 += 32) {
        union { v16h v; v8h h[2]; } ua;
        ua.h[0] = *(const v8h*)(aRow + k0);
        ua.h[1] = *(const v8h*)(aRow + k0 + 8);
        #pragma unroll
        for (int tn = 0; tn < NT; ++tn) {
            v16h bfrag = *(const v16h*)(wRow[tn] + k0);
            acc[tn] = __builtin_amdgcn_wmma_f32_16x16x32_f16(
                        false, ua.v, false, bfrag, (short)0, acc[tn], false, false);
        }
    }

    #pragma unroll
    for (int tn = 0; tn < NT; ++tn) {
        const int col = (ctBase + 2 * tn) * 16 + rlow;
        const float bias = Bias[col];
        if (!last) {
            #pragma unroll
            for (int e = 0; e < 8; ++e) {
                float val = fmaxf(acc[tn][e] + bias, 0.0f);   // ReLU
                nxt[(size_t)(rowBase + e + 8 * hi) * doutP + col] = (f16)val;
            }
        } else {
            float* fout = (float*)nxt;                        // f32 staging
            #pragma unroll
            for (int e = 0; e < 8; ++e)
                fout[(size_t)(rowBase + e + 8 * hi) * doutP + col] = acc[tn][e] + bias;
        }
    }
}

// ---------------------------------------------------------------------------
// Fused MLP over a 64-row tile. 256 threads = 8 wave32.
//   wave w: rowBlk = w&3 (16 rows), colPhase = w>>2 (interleaved col tiles).
// Activations ping-pong in LDS (fp16, compact stride = dinP).
// Weights stream from global (L2 resident) as WMMA B fragments.
// Last layer accumulates f32 into LDS; epilogue writes/normalizes outputs.
// ---------------------------------------------------------------------------
__launch_bounds__(256)
__global__ void mlp_fused_kernel(const float* __restrict__ pin, int pinDim,
                                 const float* __restrict__ tptr,
                                 const f16* __restrict__ wsW,
                                 const float* __restrict__ wsB,
                                 NetDesc net,
                                 int realIn,          // 61 or 121
                                 float* __restrict__ outp,
                                 int doNorm)
{
    __shared__ __align__(32) f16 bufA[64 * 256];   // 32 KB
    __shared__ __align__(32) f16 bufB[64 * 256];   // 32 KB

    const int tid  = threadIdx.x;
    const int wave = tid >> 5;
    const int lane = tid & 31;
    const int n0   = blockIdx.x * 64;

    // ---- featurize: posenc(p, L) ++ t, zero padded to dinP -------------
    {
        const int dinP = net.L[0].dinP;
        const float tval = tptr[0];
        const int twoD = 2 * pinDim;
        for (int idx = tid; idx < 64 * dinP; idx += 256) {
            const int r = idx / dinP;
            const int f = idx - r * dinP;
            float val;
            if (f >= realIn) {
                val = 0.0f;
            } else if (f == realIn - 1) {
                val = tval;
            } else {
                const int k   = f / twoD;
                const int rem = f - k * twoD;
                const int d   = (rem < pinDim) ? rem : rem - pinDim;
                const float p = pin[(size_t)(n0 + r) * pinDim + d];
                const float ang = exp2f((float)k) * 3.14159265358979323846f * p;
                val = (rem < pinDim) ? __sinf(ang) : __cosf(ang);
            }
            bufA[r * dinP + f] = (f16)val;
        }
    }
    __syncthreads();

    const int rowBlk   = wave & 3;
    const int colPhase = wave >> 2;
    const int hi   = lane >> 4;   // half-wave select
    const int rlow = lane & 15;
    const int rowBase = rowBlk * 16;

    f16* cur = bufA;
    f16* nxt = bufB;

    for (int li = 0; li < net.nLayers; ++li) {
        const int dinP  = net.L[li].dinP;
        const int doutP = net.L[li].doutP;
        const f16*   __restrict__ W = wsW + net.L[li].wOff;
        const float* __restrict__ B = wsB + net.L[li].bOff;
        const bool last = (li == net.nLayers - 1);

        const f16* aRow = cur + (size_t)(rowBase + rlow) * dinP + hi * 16;

        // This wave's column tiles: ct = colPhase + 2*i, i in [0, cnt)
        const int nTiles = doutP >> 4;
        const int cnt = (nTiles > colPhase) ? ((nTiles - colPhase + 1) >> 1) : 0;
        int i = 0, ct = colPhase;
        for (; i + 4 <= cnt; i += 4, ct += 8)
            mma_group<4>(aRow, W, B, nxt, dinP, doutP, ct, rowBase, hi, rlow, last);
        if (i + 2 <= cnt) {
            mma_group<2>(aRow, W, B, nxt, dinP, doutP, ct, rowBase, hi, rlow, last);
            i += 2; ct += 4;
        }
        if (i < cnt)
            mma_group<1>(aRow, W, B, nxt, dinP, doutP, ct, rowBase, hi, rlow, last);

        __syncthreads();
        f16* t_ = cur; cur = nxt; nxt = t_;
    }

    // ---- epilogue: final f32 rows live in `cur`, stride 16 --------------
    const float* fin = (const float*)cur;
    if (tid < 64) {
        const int row = tid;
        if (doNorm) {
            const float a0 = fin[row * 16 + 0];
            const float a1 = fin[row * 16 + 1];
            const float a2 = fin[row * 16 + 2];
            const float a3 = fin[row * 16 + 3];
            const float inv = rsqrtf(a0 * a0 + a1 * a1 + a2 * a2 + a3 * a3);
            float* o = outp + (size_t)(n0 + row) * 4;
            o[0] = a0 * inv; o[1] = a1 * inv; o[2] = a2 * inv; o[3] = a3 * inv;
        } else {
            float* o = outp + (size_t)(n0 + row) * 3;
            o[0] = fin[row * 16 + 0];
            o[1] = fin[row * 16 + 1];
            o[2] = fin[row * 16 + 2];
        }
    }
}

// ---------------------------------------------------------------------------
// Host side
// ---------------------------------------------------------------------------
extern "C" void kernel_launch(void* const* d_in, const int* in_sizes, int n_in,
                              void* d_out, int out_size, void* d_ws, size_t ws_size,
                              hipStream_t stream)
{
    (void)n_in; (void)ws_size; (void)out_size;
    const float* x = (const float*)d_in[0];
    const float* q = (const float*)d_in[1];
    const float* t = (const float*)d_in[2];
    const int N = in_sizes[0] / 3;          // 262144

    // Layer dims straight from the reference.
    static const int dims_x[9][2] = {
        {61,256},{256,256},{256,256},{256,256},{256,256},{256,256},
        {256,128},{128,128},{128,3}};
    static const int dims_q[9][2] = {
        {121,256},{256,256},{256,256},{256,256},{256,256},{256,256},
        {256,256},{256,128},{128,4}};

    auto padK = [](int v) { return (v + 31) & ~31; };   // K dim -> mult of 32
    auto padN = [](int v) { return (v + 15) & ~15; };   // out dim -> mult of 16

    // Workspace layout: [fp16 weights (all layers)] [f32 biases (all layers)]
    size_t wTot = 0;
    for (int i = 0; i < 9; ++i)
        wTot += (size_t)padN(dims_x[i][1]) * padK(dims_x[i][0]);
    for (int i = 0; i < 9; ++i)
        wTot += (size_t)padN(dims_q[i][1]) * padK(dims_q[i][0]);

    f16*   wsW = (f16*)d_ws;
    float* wsB = (float*)((char*)d_ws + wTot * sizeof(f16));

    NetDesc netx, netq;
    netx.nLayers = 9;
    netq.nLayers = 9;

    unsigned wOff = 0, bOff = 0;
    int inIdx = 3;
    for (int i = 0; i < 9; ++i) {
        const int din = dims_x[i][0], dout = dims_x[i][1];
        const int dinP = padK(din), doutP = padN(dout);
        netx.L[i] = {dinP, doutP, wOff, bOff};
        const float* W = (const float*)d_in[inIdx++];
        const float* b = (const float*)d_in[inIdx++];
        const int total = doutP * dinP + doutP;
        pack_layer_kernel<<<(total + 255) / 256, 256, 0, stream>>>(
            W, b, wsW + wOff, wsB + bOff, dout, din, doutP, dinP);
        wOff += (unsigned)(doutP * dinP);
        bOff += (unsigned)doutP;
    }
    for (int i = 0; i < 9; ++i) {
        const int din = dims_q[i][0], dout = dims_q[i][1];
        const int dinP = padK(din), doutP = padN(dout);
        netq.L[i] = {dinP, doutP, wOff, bOff};
        const float* W = (const float*)d_in[inIdx++];
        const float* b = (const float*)d_in[inIdx++];
        const int total = doutP * dinP + doutP;
        pack_layer_kernel<<<(total + 255) / 256, 256, 0, stream>>>(
            W, b, wsW + wOff, wsB + bOff, dout, din, doutP, dinP);
        wOff += (unsigned)(doutP * dinP);
        bOff += (unsigned)doutP;
    }

    float* out = (float*)d_out;
    const int nBlocks = N / 64;                       // 4096

    // x branch -> out[0 .. N*3)
    mlp_fused_kernel<<<nBlocks, 256, 0, stream>>>(
        x, 3, t, wsW, wsB, netx, 61, out, 0);
    // q branch (normalized) -> out[N*3 .. N*3 + N*4)
    mlp_fused_kernel<<<nBlocks, 256, 0, stream>>>(
        q, 4, t, wsW, wsB, netq, 121, out + (size_t)N * 3, 1);
}
